// TPSGeometricTransform_31533649887467
// MI455X (gfx1250) — matrix operand order repaired and checked by reading, not compiled, not run on hardware
//
#include <hip/hip_runtime.h>
#include <hip/hip_bf16.h>
#include <math.h>

typedef __attribute__((ext_vector_type(16))) _Float16 v16h;
typedef __attribute__((ext_vector_type(8)))  float    v8f;

#define HH 512
#define WWID 512
#define CC 3
#define BB 32
#define OC 8
#define OH 506
#define OW 506
#define MTOT (OH * OW)          // 256036 output positions per image
#define KDIM (CC * 49)          // 147 = 3*7*7
#define KSTEPN 5                // ceil(147/32) WMMA k-steps
#define NTILE ((MTOT + 15) / 16)
#define KNUM 16                 // TPS control points
#define NSYS 19                 // K + 3

// ---------------------------------------------------------------------------
// zero-init small workspace region (graph-capture safe)
// ---------------------------------------------------------------------------
__global__ void tps_zero_kernel(float* __restrict__ p, int n) {
    int i = blockIdx.x * blockDim.x + threadIdx.x;
    if (i < n) p[i] = 0.0f;
}

// ---------------------------------------------------------------------------
// Stage 1: conv7x7 (VALID) -> relu -> global-avg-pool, via implicit-GEMM WMMA.
// grid = (blocks_x, B). Each wave (32 lanes) owns 16-position tiles.
// A = im2col patches [16 x 160] f16, B = weights [160 x 16] f16 (cols 8..15 = 0),
// C = f32 accum. 5 x v_wmma_f32_16x16x32_f16 per tile.
//
// A-fragment gather: the K index for fragment element (s,e) depends only on
// the lane-half bit, so branching on it makes every (c,ky,kx) decode and image
// offset a compile-time constant -> plain global_load_b32 with immediate
// offset, zero div/mod VALU in the hot loop. The wave reconverges before the
// WMMAs (EXEC must be all 1s for WMMA).
// ---------------------------------------------------------------------------
__global__ void tps_conv_pool_wmma(const float* __restrict__ x,
                                   const float* __restrict__ conv_w,
                                   const float* __restrict__ conv_b,
                                   float* __restrict__ pooled) {
    const int b    = blockIdx.y;
    const int lane = threadIdx.x & 31;
    const int wid  = threadIdx.x >> 5;
    const int wavesTotal = gridDim.x * (blockDim.x >> 5);
    const int wglobal    = blockIdx.x * (blockDim.x >> 5) + wid;

    const bool hi   = lane >= 16;
    const int  ln15 = lane & 15;

    // --- B fragments (weights): batch-shared, built once per wave ----------
    // conv_w is [8,3,7,7] row-major: (oc*3+c)*49 + r == oc*147 + k  (linear!)
    v16h bf[KSTEPN];
    {
        const float* wb = conv_w + ln15 * KDIM;
        const int khBase = hi ? 8 : 0;
#pragma unroll
        for (int s = 0; s < KSTEPN; ++s) {
#pragma unroll
            for (int e = 0; e < 16; ++e) {
                const int k = s * 32 + khBase + ((e < 8) ? e : e + 8);
                float w = 0.0f;
                if (ln15 < OC && k < KDIM) w = wb[k];
                bf[s][e] = (_Float16)w;
            }
        }
    }

    const float bias = (ln15 < OC) ? conv_b[ln15] : 0.0f;
    const float* xb  = x + (size_t)b * CC * HH * WWID;
    float acc = 0.0f;

    for (int tile = wglobal; tile < NTILE; tile += wavesTotal) {
        // A row owned by this lane (both lane halves hold the same row,
        // different K ranges). Clamp invalid tail rows to position 0; their
        // C rows are masked out below.
        const int mrow  = tile * 16 + ln15;
        const bool rowok = mrow < MTOT;
        const int oy = rowok ? (mrow / OW) : 0;
        const int ox = rowok ? (mrow % OW) : 0;
        const float* xp = xb + (size_t)(oy * WWID + ox);

        // ---- build all 5 A fragments with compile-time constant offsets ---
        v16h afrag[KSTEPN];
        if (!hi) {
#pragma unroll
            for (int s = 0; s < KSTEPN; ++s) {
#pragma unroll
                for (int e = 0; e < 16; ++e) {
                    const int k = s * 32 + ((e < 8) ? e : e + 8);   // constant
                    float v = 0.0f;
                    if (k < KDIM) {
                        const int c  = k / 49, r = k % 49;          // folds
                        const int ky = r / 7,  kx = r % 7;
                        v = xp[(c * HH + ky) * WWID + kx];          // imm offs
                    }
                    afrag[s][e] = (_Float16)v;
                }
            }
        } else {
#pragma unroll
            for (int s = 0; s < KSTEPN; ++s) {
#pragma unroll
                for (int e = 0; e < 16; ++e) {
                    const int k = s * 32 + 8 + ((e < 8) ? e : e + 8); // constant
                    float v = 0.0f;
                    if (k < KDIM) {
                        const int c  = k / 49, r = k % 49;
                        const int ky = r / 7,  kx = r % 7;
                        v = xp[(c * HH + ky) * WWID + kx];
                    }
                    afrag[s][e] = (_Float16)v;
                }
            }
        }
        // reconverged here: EXEC all 1s for WMMA

        v8f cfrag = {};
#pragma unroll
        for (int s = 0; s < KSTEPN; ++s) {
            cfrag = __builtin_amdgcn_wmma_f32_16x16x32_f16(
                /*neg_a=*/false, afrag[s], /*neg_b=*/false, bf[s],
                /*c_mod=*/(short)0, cfrag, /*reuse_a=*/false, /*reuse_b=*/false);
        }

        // C layout: element r -> row tile*16 + r + (hi ? 8 : 0), col = ln15.
        // Only the very last tile has invalid rows (256036 % 16 = 4).
        if (tile < NTILE - 1) {
#pragma unroll
            for (int r = 0; r < 8; ++r) {
                float v = cfrag[r] + bias;
                acc += (v > 0.0f ? v : 0.0f);
            }
        } else {
#pragma unroll
            for (int r = 0; r < 8; ++r) {
                const int m = tile * 16 + r + (hi ? 8 : 0);
                float v = cfrag[r] + bias;
                v = v > 0.0f ? v : 0.0f;
                if (m < MTOT) acc += v;
            }
        }
    }

    if (ln15 < OC)
        atomicAdd(&pooled[b * OC + ln15], acc * (1.0f / (float)MTOT));
}

// ---------------------------------------------------------------------------
// Stage 2: FC head + TPS solve. One block of 32 threads (one per batch).
// Thread 0 builds the fixed 19x19 L and inverts it in LDS (partial pivoting --
// L's diagonal is zero, no-pivot elimination would blow up). Then each thread
// computes pred_b and W_b = L^-1 * Y_b.
// ---------------------------------------------------------------------------
__global__ void tps_solve(const float* __restrict__ pooled,
                          const float* __restrict__ fc_w,
                          const float* __restrict__ fc_b,
                          float* __restrict__ Wmat) {
    __shared__ float Ls[NSYS][NSYS];
    __shared__ float Inv[NSYS][NSYS];
    const int t = threadIdx.x;

    if (t == 0) {
        float sx[KNUM], sy[KNUM];
        for (int k = 0; k < KNUM; ++k) {
            int i = k >> 2, j = k & 3;                 // src[k] = (lin[j], lin[i])
            sx[k] = -1.0f + (float)j * (2.0f / 3.0f);
            sy[k] = -1.0f + (float)i * (2.0f / 3.0f);
        }
        for (int i = 0; i < NSYS; ++i)
            for (int j = 0; j < NSYS; ++j) Ls[i][j] = 0.0f;
        for (int i = 0; i < KNUM; ++i) {
            for (int j = 0; j < KNUM; ++j) {
                float dx = sx[i] - sx[j], dy = sy[i] - sy[j];
                float d2 = dx * dx + dy * dy;
                Ls[i][j] = d2 * __logf(d2 + 1e-6f);
            }
            Ls[i][KNUM + 0] = 1.0f;  Ls[i][KNUM + 1] = sx[i];  Ls[i][KNUM + 2] = sy[i];
            Ls[KNUM + 0][i] = 1.0f;  Ls[KNUM + 1][i] = sx[i];  Ls[KNUM + 2][i] = sy[i];
        }
        for (int i = 0; i < NSYS; ++i)
            for (int j = 0; j < NSYS; ++j) Inv[i][j] = (i == j) ? 1.0f : 0.0f;

        // Gauss-Jordan with partial pivoting
        for (int col = 0; col < NSYS; ++col) {
            int piv = col; float best = fabsf(Ls[col][col]);
            for (int r = col + 1; r < NSYS; ++r) {
                float a = fabsf(Ls[r][col]);
                if (a > best) { best = a; piv = r; }
            }
            if (piv != col) {
                for (int j = 0; j < NSYS; ++j) {
                    float tmp = Ls[col][j]; Ls[col][j] = Ls[piv][j]; Ls[piv][j] = tmp;
                    tmp = Inv[col][j]; Inv[col][j] = Inv[piv][j]; Inv[piv][j] = tmp;
                }
            }
            float d = 1.0f / Ls[col][col];
            for (int j = 0; j < NSYS; ++j) { Ls[col][j] *= d; Inv[col][j] *= d; }
            for (int r = 0; r < NSYS; ++r) {
                if (r == col) continue;
                float f = Ls[r][col];
                if (f != 0.0f) {
                    for (int j = 0; j < NSYS; ++j) {
                        Ls[r][j]  -= f * Ls[col][j];
                        Inv[r][j] -= f * Inv[col][j];
                    }
                }
            }
        }
    }
    __syncthreads();

    if (t < BB) {
        float p8[OC];
        for (int i = 0; i < OC; ++i) p8[i] = pooled[t * OC + i];
        float pred[2 * KNUM];
        for (int j = 0; j < 2 * KNUM; ++j) {
            float s = fc_b[j];
            for (int i = 0; i < OC; ++i) s += p8[i] * fc_w[j * OC + i];
            pred[j] = s;
        }
        // Y rows 0..15 = (pred_x, pred_y), rows 16..18 = 0 -> W = Inv * Y
        for (int r = 0; r < NSYS; ++r) {
            float wx = 0.0f, wy = 0.0f;
            for (int k = 0; k < KNUM; ++k) {
                float iv = Inv[r][k];
                wx += iv * pred[2 * k + 0];
                wy += iv * pred[2 * k + 1];
            }
            Wmat[(t * NSYS + r) * 2 + 0] = wx;
            Wmat[(t * NSYS + r) * 2 + 1] = wy;
        }
    }
}

// ---------------------------------------------------------------------------
// Stage 3: fused TPS grid mapping + bilinear grid_sample (memory-bound).
// Thread = one output pixel; U[16] computed once (batch-invariant: 16 logs),
// then loop over all 32 batch images. Solve coefficients staged in LDS.
// ---------------------------------------------------------------------------
__global__ void tps_warp_sample(const float* __restrict__ x,
                                const float* __restrict__ Wmat,
                                float* __restrict__ out) {
    __shared__ float Ws[BB * NSYS * 2];
    for (int i = threadIdx.x; i < BB * NSYS * 2; i += blockDim.x)
        Ws[i] = Wmat[i];
    __syncthreads();

    const int p = blockIdx.x * blockDim.x + threadIdx.x;
    if (p >= HH * WWID) return;
    const int py = p / WWID, px = p % WWID;
    const float gx = -1.0f + (float)px * (2.0f / (float)(WWID - 1));
    const float gy = -1.0f + (float)py * (2.0f / (float)(HH - 1));

    float U[KNUM];
#pragma unroll
    for (int k = 0; k < KNUM; ++k) {
        int i = k >> 2, j = k & 3;
        float sxk = -1.0f + (float)j * (2.0f / 3.0f);
        float syk = -1.0f + (float)i * (2.0f / 3.0f);
        float dx = gx - sxk, dy = gy - syk;
        float d2 = dx * dx + dy * dy;
        U[k] = d2 * __logf(d2 + 1e-6f);
    }

    for (int b = 0; b < BB; ++b) {
        const float* Wb = &Ws[b * NSYS * 2];
        float mx = Wb[2 * 16 + 0] + gx * Wb[2 * 17 + 0] + gy * Wb[2 * 18 + 0];
        float my = Wb[2 * 16 + 1] + gx * Wb[2 * 17 + 1] + gy * Wb[2 * 18 + 1];
#pragma unroll
        for (int k = 0; k < KNUM; ++k) {
            mx += U[k] * Wb[2 * k + 0];
            my += U[k] * Wb[2 * k + 1];
        }
        // align_corners=False mapping
        float ix = ((mx + 1.0f) * (float)WWID - 1.0f) * 0.5f;
        float iy = ((my + 1.0f) * (float)HH   - 1.0f) * 0.5f;
        float x0f = floorf(ix), y0f = floorf(iy);
        int   x0 = (int)x0f,    y0 = (int)y0f;
        float wx1 = ix - x0f,   wy1 = iy - y0f;
        float wx0 = 1.0f - wx1, wy0 = 1.0f - wy1;

        bool vx0 = (x0 >= 0) && (x0 < WWID);
        bool vx1 = (x0 + 1 >= 0) && (x0 + 1 < WWID);
        bool vy0 = (y0 >= 0) && (y0 < HH);
        bool vy1 = (y0 + 1 >= 0) && (y0 + 1 < HH);
        int xc0 = min(max(x0, 0), WWID - 1), xc1 = min(max(x0 + 1, 0), WWID - 1);
        int yc0 = min(max(y0, 0), HH - 1),   yc1 = min(max(y0 + 1, 0), HH - 1);
        float w00 = wy0 * wx0 * ((vy0 && vx0) ? 1.0f : 0.0f);
        float w01 = wy0 * wx1 * ((vy0 && vx1) ? 1.0f : 0.0f);
        float w10 = wy1 * wx0 * ((vy1 && vx0) ? 1.0f : 0.0f);
        float w11 = wy1 * wx1 * ((vy1 && vx1) ? 1.0f : 0.0f);

#pragma unroll
        for (int c = 0; c < CC; ++c) {
            const float* img = x + (size_t)(b * CC + c) * HH * WWID;
            float v = w00 * img[yc0 * WWID + xc0] + w01 * img[yc0 * WWID + xc1]
                    + w10 * img[yc1 * WWID + xc0] + w11 * img[yc1 * WWID + xc1];
            out[((size_t)(b * CC + c) * HH + py) * WWID + px] = v;
        }
    }
}

// ---------------------------------------------------------------------------
extern "C" void kernel_launch(void* const* d_in, const int* in_sizes, int n_in,
                              void* d_out, int out_size, void* d_ws, size_t ws_size,
                              hipStream_t stream) {
    (void)in_sizes; (void)n_in; (void)out_size; (void)ws_size;

    const float* x      = (const float*)d_in[0];   // [32,3,512,512]
    const float* conv_w = (const float*)d_in[1];   // [8,3,7,7]
    const float* conv_b = (const float*)d_in[2];   // [8]
    const float* fc_w   = (const float*)d_in[3];   // [32,8]
    const float* fc_b   = (const float*)d_in[4];   // [32]
    float* out = (float*)d_out;                    // [32,3,512,512]

    float* pooled = (float*)d_ws;                  // BB*OC      = 256 floats
    float* Wmat   = pooled + BB * OC;              // BB*NSYS*2  = 1216 floats

    tps_zero_kernel<<<1, 256, 0, stream>>>(pooled, BB * OC);

    tps_conv_pool_wmma<<<dim3(512, BB), 256, 0, stream>>>(x, conv_w, conv_b, pooled);

    tps_solve<<<1, 32, 0, stream>>>(pooled, fc_w, fc_b, Wmat);

    tps_warp_sample<<<(HH * WWID + 255) / 256, 256, 0, stream>>>(x, Wmat, out);
}
